// CliqueGNN_15324443312754
// MI455X (gfx1250) — compile-verified, matrix operand order checked by reading, not run on hardware
//
#include <hip/hip_runtime.h>
#include <hip/hip_bf16.h>

typedef _Float16 half_t;
typedef __attribute__((ext_vector_type(16))) _Float16 v16h;
typedef __attribute__((ext_vector_type(8)))  _Float16 v8h;
typedef __attribute__((ext_vector_type(8)))  float    v8f;

#define WMMA(a,b,c) __builtin_amdgcn_wmma_f32_16x16x32_f16(false,(a),false,(b),(short)0,(c),false,false)

#define GN   1024
#define GE   500000
#define GTIL (GE/16)      // 31250 exact
#define GEPS 1e-5f

// 16-bit A/B fragment element->K mapping (ISA 7.12.2, 16-bit A 16x32):
// lane<16: elems 0..7 -> K 0..7, elems 8..15 -> K 16..23 ; lane>=16: +8
__device__ __forceinline__ int kf16(int e, int lane) {
    return ((e < 8) ? e : e + 8) + ((lane & 16) ? 8 : 0);
}

// Build one v16h A fragment from two contiguous 8-half runs (global memory).
__device__ __forceinline__ v16h a_frag_f16(const half_t* __restrict__ row, int koff) {
    v8h lo = *(const v8h*)(row + koff);       // K koff .. koff+7
    v8h hi = *(const v8h*)(row + koff + 16);  // K koff+16 .. koff+23
    return __builtin_shufflevector(lo, hi, 0, 1, 2, 3, 4, 5, 6, 7,
                                           8, 9, 10, 11, 12, 13, 14, 15);
}

// ---------------- utility ----------------
__global__ void k_zero(float* __restrict__ p, int n) {
    int i = blockIdx.x * blockDim.x + threadIdx.x;
    if (i < n) p[i] = 0.f;
}

// ---------------- weight packing into WMMA B fragments ----------------
// One wave per 32(K)x16(N) fragment. Fragment stored as [lane][16 halves].
__global__ void k_pack_w(const float* __restrict__ W, half_t* __restrict__ dst,
                         int K, int Nmat, int ksteps, int nfrag) {
    int wave = (blockIdx.x * blockDim.x + threadIdx.x) >> 5;
    int lane = threadIdx.x & 31;
    if (wave >= nfrag) return;
    int nt = wave / ksteps, ks = wave - nt * ksteps;
    int n  = nt * 16 + (lane & 15);
    v16h v;
#pragma unroll
    for (int e = 0; e < 16; e++) {
        int k = ks * 32 + kf16(e, lane);
        v[e] = (half_t)((k < K) ? W[(size_t)k * Nmat + n] : 0.f);
    }
    *(v16h*)(dst + (size_t)wave * 512 + lane * 16) = v;
}

// ---------------- degrees ----------------
__global__ void k_deg(const int* __restrict__ dste, float* __restrict__ degf, int E) {
    int e = blockIdx.x * blockDim.x + threadIdx.x;
    if (e < E) atomicAdd(&degf[dste[e]], 1.f);
}
__global__ void k_deg_fin(const float* __restrict__ degf, float* __restrict__ isq,
                          float* __restrict__ idg) {
    int n = blockIdx.x * blockDim.x + threadIdx.x;
    if (n < GN) {
        float d = degf[n] + 1.f;
        isq[n] = rsqrtf(d);
        idg[n] = 1.f / d;
    }
}

// ---------------- embeddings ----------------
__global__ void k_node_emb(const float* __restrict__ w, const float* __restrict__ b,
                           float* __restrict__ x, half_t* __restrict__ xh) {
    int i = blockIdx.x * blockDim.x + threadIdx.x;  // GN*64
    if (i < GN * 64) {
        int n = i >> 6, h = i & 63;
        float v = (float)n * w[h] + b[h];
        x[i]  = v;
        xh[i] = (half_t)v;
    }
}
__global__ void k_edge_emb(const float* __restrict__ attr, const float* __restrict__ w,
                           const float* __restrict__ b, half_t* __restrict__ ef, int E) {
    int e = blockIdx.x * blockDim.x + threadIdx.x;
    if (e >= E) return;
    float a0 = attr[e * 3 + 0], a1 = attr[e * 3 + 1], a2 = attr[e * 3 + 2];
    half_t* o = ef + (size_t)e * 64;
#pragma unroll
    for (int c = 0; c < 4; c++) {
        v16h v;
#pragma unroll
        for (int j = 0; j < 16; j++) {
            int h = c * 16 + j;
            v[j] = (half_t)(a0 * w[h] + a1 * w[64 + h] + a2 * w[128 + h] + b[h]);
        }
        *(v16h*)(o + c * 16) = v;
    }
}

// ---------------- GCN: xw = x @ W (1024x64x64 via WMMA, f16 node mirror) ----------------
__global__ void k_gcn_mm(const half_t* __restrict__ xh, const half_t* __restrict__ wg,
                         float* __restrict__ xw) {
    int wave = (blockIdx.x * blockDim.x + threadIdx.x) >> 5;
    int lane = threadIdx.x & 31;
    int m0 = wave * 16;
    if (m0 >= GN) return;
    int m = lane & 15;
    int koff = (lane & 16) ? 8 : 0;
    const half_t* xr = xh + (size_t)(m0 + m) * 64;
    v16h a[2];
#pragma unroll
    for (int ks = 0; ks < 2; ks++) a[ks] = a_frag_f16(xr, ks * 32 + koff);
#pragma unroll
    for (int nt = 0; nt < 4; nt++) {
        v8f c = {};
#pragma unroll
        for (int ks = 0; ks < 2; ks++) {
            v16h b = *(const v16h*)(wg + (size_t)(nt * 2 + ks) * 512 + lane * 16);
            c = WMMA(a[ks], b, c);
        }
        int n = nt * 16 + (lane & 15);
        int mr = (lane & 16) ? 8 : 0;
#pragma unroll
        for (int j = 0; j < 8; j++) xw[(size_t)(m0 + mr + j) * 64 + n] = c[j];
    }
}

// ---------------- GCN scatter: agg[dst] += xw[src] * coef ----------------
__global__ void k_gcn_scatter(const int* __restrict__ srce, const int* __restrict__ dste,
                              const float* __restrict__ isq, const float* __restrict__ xw,
                              float* __restrict__ agg, int E) {
    int e = blockIdx.x * blockDim.x + threadIdx.x;
    if (e >= E) return;
    int s = srce[e], d = dste[e];
    float cf = isq[s] * isq[d];
    const float* xr = xw + (size_t)s * 64;
    float* ar = agg + (size_t)d * 64;
#pragma unroll
    for (int h = 0; h < 64; h++) atomicAdd(&ar[h], xr[h] * cf);
}

// ---------------- GCN post: self-loop + bias + BN(N) + ReLU (f32 + f16 mirror out) ----------------
__global__ __launch_bounds__(1024) void k_gcn_post(
    float* __restrict__ agg, const float* __restrict__ xw, const float* __restrict__ idg,
    const float* __restrict__ gb, const float* __restrict__ gamma,
    const float* __restrict__ beta, float* __restrict__ xnew, half_t* __restrict__ xhnew) {
    __shared__ float ps[16][64], pq[16][64], sc[64], sh[64];
    int t = threadIdx.x, col = t & 63, grp = t >> 6;  // 16 row-groups x 64 cols
    float s = 0.f, q = 0.f;
    for (int r = grp; r < GN; r += 16) {
        size_t i = (size_t)r * 64 + col;
        float v = agg[i] + xw[i] * idg[r] + gb[col];
        agg[i] = v;
        s += v;
        q += v * v;
    }
    ps[grp][col] = s; pq[grp][col] = q;
    __syncthreads();
    if (t < 64) {
        float S = 0.f, Q = 0.f;
#pragma unroll
        for (int g = 0; g < 16; g++) { S += ps[g][t]; Q += pq[g][t]; }
        float mu = S * (1.f / GN);
        float var = Q * (1.f / GN) - mu * mu;
        float scale = gamma[t] * rsqrtf(var + GEPS);
        sc[t] = scale; sh[t] = beta[t] - mu * scale;
    }
    __syncthreads();
    for (int r = grp; r < GN; r += 16) {
        size_t i = (size_t)r * 64 + col;
        float v = fmaxf(agg[i] * sc[col] + sh[col], 0.f);
        xnew[i]  = v;
        xhnew[i] = (half_t)v;
    }
}

// ---------------- fused EdgeBlock: 3 chained WMMA GEMMs per 16-edge tile ----------------
__global__ __launch_bounds__(256) void k_edge_block(
    const int* __restrict__ srce, const int* __restrict__ dste,
    const half_t* __restrict__ xh, const half_t* __restrict__ ef,
    const half_t* __restrict__ wn, const float* __restrict__ bnb,
    const half_t* __restrict__ we, const float* __restrict__ beb,
    const half_t* __restrict__ wc, const float* __restrict__ bcb,
    half_t* __restrict__ comb, float* __restrict__ gsum, float* __restrict__ gsq,
    int ntiles) {
    __shared__ __attribute__((aligned(64))) half_t ldsA[8][16][128];  // per-wave staging
    int lane = threadIdx.x & 31;
    int wslot = threadIdx.x >> 5;
    int gw = (blockIdx.x * blockDim.x + threadIdx.x) >> 5;
    int nw = (gridDim.x * blockDim.x) >> 5;
    int m = lane & 15;
    int mr = (lane & 16) ? 8 : 0;
    int koff = (lane & 16) ? 8 : 0;

    // per-lane BN stats accumulators: column n = nt*16 + (lane&15), rows mr..mr+7
    float sA[4] = {0.f, 0.f, 0.f, 0.f};
    float qA[4] = {0.f, 0.f, 0.f, 0.f};

    for (int t = gw; t < ntiles; t += nw) {
        int e0 = t * 16;
        // prefetch next tile's ef row (global_prefetch_b8)
        int tn = t + nw;
        if (tn < ntiles) __builtin_prefetch(ef + (size_t)(tn * 16 + m) * 64, 0, 1);

        int es = srce[e0 + m], ed = dste[e0 + m];
        const half_t* xs = xh + (size_t)es * 64;
        const half_t* xd = xh + (size_t)ed * 64;

        // A for nf GEMM: [16 edges][128] = [xh[src] | xh[dst]] -> 4 k-steps (pure f16)
        v16h a[4];
#pragma unroll
        for (int ks = 0; ks < 2; ks++) a[ks]     = a_frag_f16(xs, ks * 32 + koff);
#pragma unroll
        for (int ks = 0; ks < 2; ks++) a[2 + ks] = a_frag_f16(xd, ks * 32 + koff);

        // nf = A @ Wn + bn -> LDS cols 0..63
#pragma unroll
        for (int nt = 0; nt < 4; nt++) {
            v8f c = {};
#pragma unroll
            for (int ks = 0; ks < 4; ks++) {
                v16h b = *(const v16h*)(wn + (size_t)(nt * 4 + ks) * 512 + lane * 16);
                c = WMMA(a[ks], b, c);
            }
            int n = nt * 16 + (lane & 15);
            float bias = bnb[n];
#pragma unroll
            for (int j = 0; j < 8; j++) ldsA[wslot][mr + j][n] = (half_t)(c[j] + bias);
        }

        // A for ee GEMM from ef (two b128 loads per fragment)
        const half_t* er = ef + (size_t)(e0 + m) * 64;
        v16h ae[2];
#pragma unroll
        for (int ks = 0; ks < 2; ks++) ae[ks] = a_frag_f16(er, ks * 32 + koff);

        // ee = ef @ We + be -> LDS cols 64..127
#pragma unroll
        for (int nt = 0; nt < 4; nt++) {
            v8f c = {};
#pragma unroll
            for (int ks = 0; ks < 2; ks++) {
                v16h b = *(const v16h*)(we + (size_t)(nt * 2 + ks) * 512 + lane * 16);
                c = WMMA(ae[ks], b, c);
            }
            int n = nt * 16 + (lane & 15);
            float bias = beb[n];
#pragma unroll
            for (int j = 0; j < 8; j++) ldsA[wslot][mr + j][64 + n] = (half_t)(c[j] + bias);
        }

        // A for comb GEMM = [nf | ee] from LDS (two ds_load_b128 per fragment)
        v16h ac[4];
#pragma unroll
        for (int ks = 0; ks < 4; ks++) {
            v8h lo = *(const v8h*)&ldsA[wslot][m][ks * 32 + koff];
            v8h hi = *(const v8h*)&ldsA[wslot][m][ks * 32 + koff + 16];
            ac[ks] = __builtin_shufflevector(lo, hi, 0, 1, 2, 3, 4, 5, 6, 7,
                                                     8, 9, 10, 11, 12, 13, 14, 15);
        }

        // comb = [nf|ee] @ Wc + bc ; per-lane stats + f16 result back into LDS cols 0..63
#pragma unroll
        for (int nt = 0; nt < 4; nt++) {
            v8f c = {};
#pragma unroll
            for (int ks = 0; ks < 4; ks++) {
                v16h b = *(const v16h*)(wc + (size_t)(nt * 4 + ks) * 512 + lane * 16);
                c = WMMA(ac[ks], b, c);
            }
            int n = nt * 16 + (lane & 15);
            float bias = bcb[n];
            float sv = 0.f, qv = 0.f;
#pragma unroll
            for (int j = 0; j < 8; j++) {
                float v = c[j] + bias;
                sv += v; qv += v * v;
                ldsA[wslot][mr + j][n] = (half_t)v;
            }
            sA[nt] += sv;
            qA[nt] += qv;
        }

        // coalesced f16 store of the 16x64 comb tile
        int r = lane & 15, cc = (lane >> 4) * 32;
        v16h d0 = *(v16h*)&ldsA[wslot][r][cc];
        v16h d1 = *(v16h*)&ldsA[wslot][r][cc + 16];
        *(v16h*)(comb + (size_t)(e0 + r) * 64 + cc)      = d0;
        *(v16h*)(comb + (size_t)(e0 + r) * 64 + cc + 16) = d1;
    }

    // flush BN stats: combine half-waves (rows 0..7 + 8..15), one atomic per column
#pragma unroll
    for (int nt = 0; nt < 4; nt++) {
        sA[nt] += __shfl_xor(sA[nt], 16, 32);
        qA[nt] += __shfl_xor(qA[nt], 16, 32);
    }
    if (lane < 16) {
#pragma unroll
        for (int nt = 0; nt < 4; nt++) {
            atomicAdd(&gsum[nt * 16 + lane], sA[nt]);
            atomicAdd(&gsq[nt * 16 + lane],  qA[nt]);
        }
    }
}

// ---------------- BN(E) finalize + apply ----------------
__global__ void k_bn_final(const float* __restrict__ s, const float* __restrict__ q,
                           const float* __restrict__ gamma, const float* __restrict__ beta,
                           float* __restrict__ scale, float* __restrict__ shift, float invE) {
    int i = threadIdx.x;  // 64
    float mu = s[i] * invE;
    float var = q[i] * invE - mu * mu;
    float sc = gamma[i] * rsqrtf(var + GEPS);
    scale[i] = sc; shift[i] = beta[i] - mu * sc;
}
__global__ void k_bn_apply(const half_t* __restrict__ comb, const float* __restrict__ scale,
                           const float* __restrict__ shift, half_t* __restrict__ ef, int E) {
    int e = blockIdx.x * blockDim.x + threadIdx.x;
    if (e >= E) return;
    const half_t* ci = comb + (size_t)e * 64;
    half_t* eo = ef + (size_t)e * 64;
#pragma unroll
    for (int c = 0; c < 4; c++) {
        v16h d = *(const v16h*)(ci + c * 16);
        v16h o;
#pragma unroll
        for (int j = 0; j < 16; j++) {
            float v = (float)d[j] * scale[c * 16 + j] + shift[c * 16 + j];
            o[j] = (half_t)fmaxf(v, 0.f);
        }
        *(v16h*)(eo + c * 16) = o;
    }
}

// ---------------- policy head: relu(ef@P1+b1)@P2+b2 fused ----------------
__global__ __launch_bounds__(256) void k_policy(
    const half_t* __restrict__ ef, const half_t* __restrict__ wp1,
    const float* __restrict__ p1b, const float* __restrict__ p2w,
    const float* __restrict__ p2b, float* __restrict__ out, int ntiles) {
    int lane = threadIdx.x & 31;
    int gw = (blockIdx.x * blockDim.x + threadIdx.x) >> 5;
    int nw = (gridDim.x * blockDim.x) >> 5;
    int m = lane & 15;
    int mr = (lane & 16) ? 8 : 0;
    int koff = (lane & 16) ? 8 : 0;
    float bias2 = p2b[0];
    for (int t = gw; t < ntiles; t += nw) {
        int e0 = t * 16;
        int tn = t + nw;
        if (tn < ntiles) __builtin_prefetch(ef + (size_t)(tn * 16 + m) * 64, 0, 1);

        const half_t* er = ef + (size_t)(e0 + m) * 64;
        v16h a[2];
#pragma unroll
        for (int ks = 0; ks < 2; ks++) a[ks] = a_frag_f16(er, ks * 32 + koff);

        float acc[8] = {0.f, 0.f, 0.f, 0.f, 0.f, 0.f, 0.f, 0.f};
#pragma unroll
        for (int nt = 0; nt < 16; nt++) {
            v8f c = {};
#pragma unroll
            for (int ks = 0; ks < 2; ks++) {
                v16h b = *(const v16h*)(wp1 + (size_t)(nt * 2 + ks) * 512 + lane * 16);
                c = WMMA(a[ks], b, c);
            }
            int n = nt * 16 + (lane & 15);
            float b1 = p1b[n], w2 = p2w[n];
#pragma unroll
            for (int j = 0; j < 8; j++) acc[j] += fmaxf(c[j] + b1, 0.f) * w2;
        }
#pragma unroll
        for (int off = 1; off < 16; off <<= 1)
#pragma unroll
            for (int j = 0; j < 8; j++) acc[j] += __shfl_xor(acc[j], off, 32);
        if ((lane & 15) == 0) {
#pragma unroll
            for (int j = 0; j < 8; j++) out[e0 + mr + j] = acc[j] + bias2;
        }
    }
}

// ---------------- value head ----------------
__global__ void k_value(const float* __restrict__ x, const float* __restrict__ v1w,
                        const float* __restrict__ v1b, const float* __restrict__ v2w,
                        const float* __restrict__ v2b, float* __restrict__ out) {
    __shared__ float pooled[64], hv[32];
    int t = threadIdx.x;  // 64 threads
    if (t < 64) {
        float s = 0.f;
        for (int r = 0; r < GN; r++) s += x[(size_t)r * 64 + t];
        pooled[t] = s * (1.f / GN);
    }
    __syncthreads();
    if (t < 32) {
        float s = v1b[t];
        for (int h = 0; h < 64; h++) s += pooled[h] * v1w[h * 32 + t];
        hv[t] = fmaxf(s, 0.f);
    }
    __syncthreads();
    if (t == 0) {
        float s = v2b[0];
        for (int j = 0; j < 32; j++) s += hv[j] * v2w[j];
        out[0] = tanhf(s);
    }
}

// ---------------- launch ----------------
extern "C" void kernel_launch(void* const* d_in, const int* in_sizes, int n_in,
                              void* d_out, int out_size, void* d_ws, size_t ws_size,
                              hipStream_t stream) {
    const int*   eidx  = (const int*)d_in[0];
    const int*   esrc  = eidx;
    const int*   edst  = eidx + GE;
    const float* eattr = (const float*)d_in[1];
    const float* nw    = (const float*)d_in[2];
    const float* nb    = (const float*)d_in[3];
    const float* eew   = (const float*)d_in[4];
    const float* eeb   = (const float*)d_in[5];
    const float* gcnw  = (const float*)d_in[6];
    const float* gcnb  = (const float*)d_in[7];
    const float* ggam  = (const float*)d_in[8];
    const float* gbet  = (const float*)d_in[9];
    const float* ebnw  = (const float*)d_in[10];
    const float* ebnb  = (const float*)d_in[11];
    const float* ebew  = (const float*)d_in[12];
    const float* ebeb  = (const float*)d_in[13];
    const float* ebcw  = (const float*)d_in[14];
    const float* ebcb  = (const float*)d_in[15];
    const float* egam  = (const float*)d_in[16];
    const float* ebet  = (const float*)d_in[17];
    const float* p1w   = (const float*)d_in[18];
    const float* p1b   = (const float*)d_in[19];
    const float* p2w   = (const float*)d_in[20];
    const float* p2b   = (const float*)d_in[21];
    const float* v1w   = (const float*)d_in[22];
    const float* v1b   = (const float*)d_in[23];
    const float* v2w   = (const float*)d_in[24];
    const float* v2b   = (const float*)d_in[25];
    float* out = (float*)d_out;

    // workspace carve-out (256B aligned)
    char* base = (char*)d_ws;
    size_t off = 0;
    auto alloc = [&](size_t bytes) -> char* {
        char* p = base + off;
        off += (bytes + 255) & ~(size_t)255;
        return p;
    };
    half_t* ef   = (half_t*)alloc((size_t)GE * 64 * 2);
    half_t* comb = (half_t*)alloc((size_t)GE * 64 * 2);
    float* xA   = (float*)alloc((size_t)GN * 64 * 4);
    float* xB   = (float*)alloc((size_t)GN * 64 * 4);
    half_t* xhA = (half_t*)alloc((size_t)GN * 64 * 2);
    half_t* xhB = (half_t*)alloc((size_t)GN * 64 * 2);
    float* xw   = (float*)alloc((size_t)GN * 64 * 4);
    float* agg  = (float*)alloc((size_t)GN * 64 * 4);
    float* degf = (float*)alloc(GN * 4);
    float* isq  = (float*)alloc(GN * 4);
    float* idg  = (float*)alloc(GN * 4);
    float* gsum = (float*)alloc(64 * 4);
    float* gsq  = (float*)alloc(64 * 4);
    float* bsc  = (float*)alloc(64 * 4);
    float* bsh  = (float*)alloc(64 * 4);
    half_t* wn  = (half_t*)alloc((size_t)2 * 16 * 512 * 2);
    half_t* we  = (half_t*)alloc((size_t)2 * 8  * 512 * 2);
    half_t* wc  = (half_t*)alloc((size_t)2 * 16 * 512 * 2);
    half_t* wg  = (half_t*)alloc((size_t)2 * 8  * 512 * 2);
    half_t* wp1 = (half_t*)alloc((size_t)32 * 512 * 2);

    // pack weights into WMMA B fragments
    for (int l = 0; l < 2; l++) {
        k_pack_w<<<2, 256, 0, stream>>>(ebnw + l * 8192, wn + l * 16 * 512, 128, 64, 4, 16);
        k_pack_w<<<1, 256, 0, stream>>>(ebew + l * 4096, we + l * 8 * 512,  64,  64, 2, 8);
        k_pack_w<<<2, 256, 0, stream>>>(ebcw + l * 8192, wc + l * 16 * 512, 128, 64, 4, 16);
        k_pack_w<<<1, 256, 0, stream>>>(gcnw + l * 4096, wg + l * 8 * 512,  64,  64, 2, 8);
    }
    k_pack_w<<<4, 256, 0, stream>>>(p1w, wp1, 64, 256, 2, 32);

    // degrees + embeddings
    k_zero<<<(GN + 255) / 256, 256, 0, stream>>>(degf, GN);
    k_deg<<<(GE + 255) / 256, 256, 0, stream>>>(edst, degf, GE);
    k_deg_fin<<<(GN + 255) / 256, 256, 0, stream>>>(degf, isq, idg);
    k_node_emb<<<(GN * 64 + 255) / 256, 256, 0, stream>>>(nw, nb, xA, xhA);
    k_edge_emb<<<(GE + 255) / 256, 256, 0, stream>>>(eattr, eew, eeb, ef, GE);

    float*  xold  = xA;
    float*  xnew  = xB;
    half_t* xhold = xhA;
    half_t* xhnew = xhB;
    for (int l = 0; l < 2; l++) {
        // GCN conv
        k_zero<<<(GN * 64 + 255) / 256, 256, 0, stream>>>(agg, GN * 64);
        k_gcn_mm<<<16, 128, 0, stream>>>(xhold, wg + l * 8 * 512, xw);
        k_gcn_scatter<<<(GE + 255) / 256, 256, 0, stream>>>(esrc, edst, isq, xw, agg, GE);
        k_gcn_post<<<1, 1024, 0, stream>>>(agg, xw, idg, gcnb + l * 64, ggam + l * 64,
                                           gbet + l * 64, xnew, xhnew);
        // edge block (uses OLD x, f16 mirror)
        k_zero<<<1, 64, 0, stream>>>(gsum, 64);
        k_zero<<<1, 64, 0, stream>>>(gsq, 64);
        k_edge_block<<<512, 256, 0, stream>>>(esrc, edst, xhold, ef,
                                              wn + l * 16 * 512, ebnb + l * 64,
                                              we + l * 8 * 512,  ebeb + l * 64,
                                              wc + l * 16 * 512, ebcb + l * 64,
                                              comb, gsum, gsq, GTIL);
        k_bn_final<<<1, 64, 0, stream>>>(gsum, gsq, egam + l * 64, ebet + l * 64,
                                         bsc, bsh, 1.f / (float)GE);
        k_bn_apply<<<(GE + 255) / 256, 256, 0, stream>>>(comb, bsc, bsh, ef, GE);
        // swap node buffers (f32 + f16 mirror together)
        float*  tf = xold;  xold  = xnew;  xnew  = tf;
        half_t* th = xhold; xhold = xhnew; xhnew = th;
    }

    // heads
    k_policy<<<512, 256, 0, stream>>>(ef, wp1, p1b, p2w, p2b, out, GTIL);
    k_value<<<1, 64, 0, stream>>>(xold, v1w, v1b, v2w, v2b, out + GE);
}
